// HydraMoDBlockWrapper_59657095741549
// MI455X (gfx1250) — matrix operand order repaired and checked by reading, not compiled, not run on hardware
//
#include <hip/hip_runtime.h>

// ---------------------------------------------------------------------------
// MoD block for MI455X (gfx1250, wave32, WMMA bf16 16x16x32, f32 accumulate)
//   B=4, L=4096, D=2048, DF=8192, k = L/2 = 2048
// Pipeline:
//   1) zero d_out
//   2) router scores  s[b,l] = x[b,l,:] . w_r + b_r          (wave per token)
//   3) per-batch bitonic top-k (exact set, jax tie-break)    (block per batch)
//   4) one-time bf16 packs in ws: w1T[DF,D], w2T[D,DF], gathered xsel[B,k,D]
//   5) fused FFN: per 32-token tile, chunked over DF=512:
//        H = gelu(Xsel @ W1 + b1)  -> bf16 in LDS (padded rows)
//        Out += H @ W2             -> 16x 16x16 f32 acc tiles per wave
//      final: + b2, scatter rows to d_out[b, idx, :]
// Inner GEMM loops are pure global_load_b128/ds_load_b128 + v_wmma (no cvt).
// ---------------------------------------------------------------------------

#define B_  4
#define L_  4096
#define D_  2048
#define DF_ 8192
#define KSEL_ 2048              // L * 0.5
#define MTILE 32                // tokens per workgroup
#define FCHUNK 512              // DF chunk per iteration
#define HSTR 520                // LDS H row stride (bf16 elems): 4-dword skew

typedef __attribute__((ext_vector_type(16))) __bf16 bf16x16;
typedef __attribute__((ext_vector_type(8)))  float  floatx8;

union Frag {
    bf16x16      v;
    unsigned int u[8];
};

// ---- helpers --------------------------------------------------------------

// f32 -> bf16 (RNE); let the backend pick a native cvt if gfx1250 has one.
__device__ __forceinline__ unsigned short f2bf(float f) {
    union { __bf16 b; unsigned short u; } c;
    c.b = (__bf16)f;
    return c.u;
}

#if __has_builtin(__builtin_amdgcn_tanhf)
__device__ __forceinline__ float tanh_hw(float v) {
    return __builtin_amdgcn_tanhf(v);                 // v_tanh_f32
}
#else
__device__ __forceinline__ float tanh_hw(float v) { return tanhf(v); }
#endif

__device__ __forceinline__ float gelu_tanh(float v) {
    return 0.5f * v * (1.0f + tanh_hw(0.7978845608028654f * (v + 0.044715f * v * v * v)));
}

// A-fragment (16x32 bf16) from a bf16 row (global xsel or LDS H tile).
// Layout: lane (m=l&15, h=l>>4): V0..7 = K = h*16 + {0..15}
__device__ __forceinline__ void load_a_bf16(Frag& f, const unsigned short* row, int k0, int h) {
    uint4 q0 = *reinterpret_cast<const uint4*>(row + k0 + h * 16);
    uint4 q1 = *reinterpret_cast<const uint4*>(row + k0 + h * 16 + 8);
    f.u[0] = q0.x; f.u[1] = q0.y; f.u[2] = q0.z; f.u[3] = q0.w;
    f.u[4] = q1.x; f.u[5] = q1.y; f.u[6] = q1.z; f.u[7] = q1.w;
}

// B-fragment (32x16 bf16) from a transposed bf16 weight row (column of W).
// Layout: lane (n=l&15, h=l>>4): V0..7 = K = h*16 + {0..15} -> one 32B run.
__device__ __forceinline__ void load_b_bf16(Frag& f, const unsigned short* row, int k0, int h) {
    const uint4* p = reinterpret_cast<const uint4*>(row + k0 + h * 16);
    uint4 q0 = p[0], q1 = p[1];
    f.u[0] = q0.x; f.u[1] = q0.y; f.u[2] = q0.z; f.u[3] = q0.w;
    f.u[4] = q1.x; f.u[5] = q1.y; f.u[6] = q1.z; f.u[7] = q1.w;
}

// ---- kernel 1: zero output ------------------------------------------------

__global__ void k_zero(float4* out, long long n4) {
    long long i = (long long)blockIdx.x * blockDim.x + threadIdx.x;
    if (i < n4) out[i] = make_float4(0.f, 0.f, 0.f, 0.f);
}

// ---- kernel 2: router scores (one wave per token) -------------------------

__global__ void k_router(const float* __restrict__ x, const float* __restrict__ w_r,
                         const float* __restrict__ b_r, float* __restrict__ scores) {
    int token = blockIdx.x * 8 + (threadIdx.x >> 5);
    int lane  = threadIdx.x & 31;
    const float* xr = x + (long long)token * D_;
    float s = 0.f;
    for (int i = lane; i < D_; i += 32) s += xr[i] * w_r[i];
    for (int o = 16; o; o >>= 1) s += __shfl_xor(s, o, 32);
    if (lane == 0) scores[token] = s + b_r[0];
}

// ---- kernel 3: per-batch bitonic top-k (exact; ties -> lower index first) --

__global__ void k_topk(const float* __restrict__ scores, int* __restrict__ sidx) {
    __shared__ float sk[L_];
    __shared__ int   sv[L_];
    const int b = blockIdx.x, tid = threadIdx.x, nt = blockDim.x;
    for (int i = tid; i < L_; i += nt) { sk[i] = scores[b * L_ + i]; sv[i] = i; }
    for (int sz = 2; sz <= L_; sz <<= 1) {
        for (int st = sz >> 1; st > 0; st >>= 1) {
            __syncthreads();
            for (int i = tid; i < L_; i += nt) {
                int j = i ^ st;
                if (j > i) {
                    float ka = sk[i], kb = sk[j];
                    int   ia = sv[i], ib = sv[j];
                    bool first_j = (kb > ka) || (kb == ka && ib < ia); // j precedes i
                    bool asc = ((i & sz) == 0);
                    if (asc ? first_j : !first_j) {
                        sk[i] = kb; sk[j] = ka; sv[i] = ib; sv[j] = ia;
                    }
                }
            }
        }
    }
    __syncthreads();
    for (int i = tid; i < KSEL_; i += nt) sidx[b * KSEL_ + i] = sv[i];
}

// ---- kernel 4/5: transpose + convert weights to bf16 in workspace ---------

__global__ void k_w1T(const float* __restrict__ w1, unsigned short* __restrict__ w1T) {
    long long o = (long long)blockIdx.x * blockDim.x + threadIdx.x;  // o over DF*D
    if (o >= (long long)DF_ * D_) return;
    int f = (int)(o / D_), d = (int)(o % D_);
    w1T[o] = f2bf(w1[(long long)d * DF_ + f]);
}
__global__ void k_w2T(const float* __restrict__ w2, unsigned short* __restrict__ w2T) {
    long long o = (long long)blockIdx.x * blockDim.x + threadIdx.x;  // o over D*DF
    if (o >= (long long)D_ * DF_) return;
    int d = (int)(o / DF_), f = (int)(o % DF_);
    w2T[o] = f2bf(w2[(long long)f * D_ + d]);
}

// ---- kernel 6: gather selected tokens, convert to bf16 --------------------

__global__ void k_xsel(const float* __restrict__ x, const int* __restrict__ sidx,
                       unsigned short* __restrict__ xs) {
    long long o = (long long)blockIdx.x * blockDim.x + threadIdx.x;  // over B*KSEL*D
    if (o >= (long long)B_ * KSEL_ * D_) return;
    int d = (int)(o % D_);
    long long row = o / D_;                   // b*KSEL + i
    int b = (int)(row / KSEL_);
    int tok = sidx[row];
    xs[o] = f2bf(x[((long long)b * L_ + tok) * D_ + d]);
}

// ---- kernel 7: fused FFN on selected tokens -------------------------------
// grid = B * (KSEL/MTILE) = 256 blocks, 512 threads (16 wave32s).
// wave w: mi = w>>3 (M half), sg = w&7.
//   GEMM1: 4 N-subtiles of the 512-wide H chunk   (cols sg*64 + s*16)
//   GEMM2: 16 N-subtiles of the 2048-wide output  (cols sg*256 + t*16)

__global__ void __launch_bounds__(512)
k_ffn(const unsigned short* __restrict__ xs, const unsigned short* __restrict__ w1T,
      const unsigned short* __restrict__ w2T, const float* __restrict__ b1,
      const float* __restrict__ b2, const int* __restrict__ sidx,
      float* __restrict__ out) {
    __shared__ unsigned short Hs[MTILE * HSTR];   // 33,280 B, bf16 H chunk
    __shared__ int sIdxTile[MTILE];

    const int b        = blockIdx.x >> 6;           // 64 tiles per batch
    const int tileBase = (blockIdx.x & 63) * MTILE;
    const int w    = threadIdx.x >> 5;
    const int lane = threadIdx.x & 31;
    const int n16  = lane & 15;
    const int h    = lane >> 4;
    const int mi   = w >> 3;                        // M half: 0 or 1
    const int sg   = w & 7;

    if (threadIdx.x < MTILE)
        sIdxTile[threadIdx.x] = sidx[b * KSEL_ + tileBase + threadIdx.x];
    __syncthreads();

    // per-lane A row for GEMM1: pre-gathered bf16 token row
    const unsigned short* xrow =
        xs + (long long)(b * KSEL_ + tileBase + mi * 16 + n16) * D_;

    // output accumulators: 16 tiles of 16x16 f32 (8 VGPRs each)
    floatx8 oacc[16];
#pragma unroll
    for (int t = 0; t < 16; ++t) oacc[t] = (floatx8)0.f;

    for (int f0 = 0; f0 < DF_; f0 += FCHUNK) {
        // ---------------- GEMM1: H[32,512] = gelu(Xsel @ W1 + b1) ----------
        floatx8 acc[4];
#pragma unroll
        for (int s = 0; s < 4; ++s) acc[s] = (floatx8)0.f;

        const unsigned short* b1row0 = w1T + (long long)(f0 + sg * 64 + n16) * D_;
        for (int k0 = 0; k0 < D_; k0 += 32) {
            Frag a; load_a_bf16(a, xrow, k0, h);
#pragma unroll
            for (int s = 0; s < 4; ++s) {
                Frag bb; load_b_bf16(bb, b1row0 + (long long)s * 16 * D_, k0, h);
                acc[s] = __builtin_amdgcn_wmma_f32_16x16x32_bf16(
                    false, a.v, false, bb.v, (short)0, acc[s], false, false);
            }
        }
        // bias + gelu -> bf16 into LDS (row = mi*16 + 8h + r, col = sg*64+s*16+n)
#pragma unroll
        for (int s = 0; s < 4; ++s) {
            int lcol = sg * 64 + s * 16 + n16;
            float bias = b1[f0 + lcol];
#pragma unroll
            for (int r = 0; r < 8; ++r) {
                int row = mi * 16 + 8 * h + r;
                float v = gelu_tanh(acc[s][r] + bias);
                Hs[row * HSTR + lcol] = f2bf(v);
            }
        }
        __syncthreads();

        // prefetch next chunk's first W1 column rows (overlaps with GEMM2)
        if (f0 + FCHUNK < DF_)
            __builtin_prefetch(w1T + (long long)(f0 + FCHUNK + sg * 64 + n16) * D_, 0, 1);

        // ---------------- GEMM2: Out[32,2048] += H @ W2 --------------------
        const unsigned short* hrow = Hs + (mi * 16 + n16) * HSTR;
        const unsigned short* w2row0 = w2T + (long long)(sg * 256 + n16) * DF_ + f0;
        for (int kk = 0; kk < FCHUNK; kk += 32) {
            Frag a; load_a_bf16(a, hrow, kk, h);
#pragma unroll
            for (int t = 0; t < 16; ++t) {
                Frag bb; load_b_bf16(bb, w2row0 + (long long)t * 16 * DF_, kk, h);
                oacc[t] = __builtin_amdgcn_wmma_f32_16x16x32_bf16(
                    false, a.v, false, bb.v, (short)0, oacc[t], false, false);
            }
        }
        __syncthreads();   // H chunk reused next iteration
    }

    // ---------------- epilogue: + b2, scatter rows to d_out ---------------
    float* orow[8];
#pragma unroll
    for (int r = 0; r < 8; ++r) {
        int tok = sIdxTile[mi * 16 + 8 * h + r];
        orow[r] = out + ((long long)b * L_ + tok) * D_;
    }
#pragma unroll
    for (int t = 0; t < 16; ++t) {
        int col = sg * 256 + t * 16 + n16;
        float bias = b2[col];
#pragma unroll
        for (int r = 0; r < 8; ++r)
            orow[r][col] = oacc[t][r] + bias;
    }
}

// ---------------------------------------------------------------------------

extern "C" void kernel_launch(void* const* d_in, const int* in_sizes, int n_in,
                              void* d_out, int out_size, void* d_ws, size_t ws_size,
                              hipStream_t stream) {
    (void)in_sizes; (void)n_in; (void)out_size; (void)ws_size;
    const float* x   = (const float*)d_in[0];
    const float* w_r = (const float*)d_in[1];
    const float* b_r = (const float*)d_in[2];
    const float* w1  = (const float*)d_in[3];
    const float* b1  = (const float*)d_in[4];
    const float* w2  = (const float*)d_in[5];
    const float* b2  = (const float*)d_in[6];
    float* out = (float*)d_out;

    // workspace layout (bytes):
    //   w1T bf16 (32MB) | w2T bf16 (32MB) | xsel bf16 (32MB) | scores | sidx
    char* ws = (char*)d_ws;
    unsigned short* w1T  = (unsigned short*)(ws);
    unsigned short* w2T  = (unsigned short*)(ws + 33554432);
    unsigned short* xsel = (unsigned short*)(ws + 67108864);
    float*          sc   = (float*)(ws + 100663296);
    int*            sidx = (int*)(ws + 100728832);

    // 1) zero output (B*L*D floats = 8M float4)
    long long n4 = (long long)B_ * L_ * D_ / 4;
    k_zero<<<(int)((n4 + 255) / 256), 256, 0, stream>>>((float4*)out, n4);

    // 2) router scores: one wave per token, 8 waves per block
    k_router<<<(B_ * L_) / 8, 256, 0, stream>>>(x, w_r, b_r, sc);

    // 3) exact top-k per batch
    k_topk<<<B_, 256, 0, stream>>>(sc, sidx);

    // 4) weight transpose + bf16 pack
    long long nw = (long long)D_ * DF_;
    k_w1T<<<(int)((nw + 255) / 256), 256, 0, stream>>>(w1, w1T);
    k_w2T<<<(int)((nw + 255) / 256), 256, 0, stream>>>(w2, w2T);

    // 5) gather + convert selected tokens to bf16
    long long nx = (long long)B_ * KSEL_ * D_;
    k_xsel<<<(int)((nx + 255) / 256), 256, 0, stream>>>(x, sidx, xsel);

    // 6) fused FFN on selected tokens: 4 batches * 64 tiles of 32 tokens
    k_ffn<<<B_ * (KSEL_ / MTILE), 512, 0, stream>>>(xsel, w1T, w2T, b1, b2, sidx, out);
}
